// spiking_coESN_rescaled_8169027797835
// MI455X (gfx1250) — matrix operand order repaired and checked by reading, not compiled, not run on hardware
//
#include <hip/hip_runtime.h>

// ---- problem constants (match reference) ----
#define Hs 1024
#define Bs 128
#define Ls 1000
#define NBLK 64          // persistent workgroups, each owns Hs/NBLK = 16 columns
#define COLS 16
#define NTHR 256         // 8 wave32s -> 8 row-tiles of 16 covering B=128

#define DT_C 0.042f
#define LIF_TAU_M_C 20.0f
#define TAU_REF_C 0.25f
#define THETA_LIF_C 1.0f
#define THETA_RF_C 1.0f

typedef __attribute__((ext_vector_type(16))) __bf16 v16bf;
typedef __attribute__((ext_vector_type(8)))  float  v8f;
typedef __attribute__((ext_vector_type(4)))  unsigned u32x4;

union BF16Frag { v16bf v; u32x4 q[2]; };

__device__ inline unsigned short f2bf(float f) {
  union { float f; unsigned u; } c; c.f = f;
  unsigned r = c.u + 0x7FFFu + ((c.u >> 16) & 1u);   // round-to-nearest-even
  return (unsigned short)(r >> 16);
}

// Sense-reversing grid barrier over NBLK persistent workgroups (agent scope).
__device__ inline void grid_sync(unsigned* cnt, unsigned* gen) {
  __threadfence();          // release all this wave's global stores to device scope
  __syncthreads();
  if (threadIdx.x == 0) {
    unsigned g = __hip_atomic_load(gen, __ATOMIC_RELAXED, __HIP_MEMORY_SCOPE_AGENT);
    unsigned a = __hip_atomic_fetch_add(cnt, 1u, __ATOMIC_ACQ_REL, __HIP_MEMORY_SCOPE_AGENT);
    if (a == (unsigned)(NBLK - 1)) {
      __hip_atomic_store(cnt, 0u, __ATOMIC_RELAXED, __HIP_MEMORY_SCOPE_AGENT);
      __hip_atomic_fetch_add(gen, 1u, __ATOMIC_RELEASE, __HIP_MEMORY_SCOPE_AGENT);
    } else {
      while (__hip_atomic_load(gen, __ATOMIC_ACQUIRE, __HIP_MEMORY_SCOPE_AGENT) == g)
        __builtin_amdgcn_s_sleep(2);
    }
  }
  __syncthreads();
  __threadfence();          // acquire: subsequent loads see other WGs' spike stores
}

// One 16x16 output tile of (spikes[128x1024] @ W[1024x16]) accumulated over K=1024
// using v_wmma_f32_16x16x32_bf16.  Sg: global bf16 spike matrix, row-major [128][1024].
// Wt: LDS, transposed slice [COLS][1024] bf16 (Wt[n][k] = W[k][n0+n]).
__device__ inline v8f spike_matmul(const unsigned short* __restrict__ Sg,
                                   const unsigned short* Wt,
                                   int mA, int half, int nl, v8f acc) {
  const unsigned short* arow = Sg + (size_t)mA * Hs;
  const unsigned short* brow = Wt + nl * Hs + 8 * half;
#pragma unroll 4
  for (int kt = 0; kt < Hs / 32; ++kt) {
    const int kb = kt * 32;
    BF16Frag a, b;
    // A (16x32 bf16): lanes 0-15 M=lane K=[kb..kb+7]+[kb+16..kb+23];
    //                 lanes16-31 same M, K offset +8 (ISA 16-bit A layout)
    const u32x4* ap = (const u32x4*)(arow + kb + 8 * half);
    a.q[0] = ap[0];
    a.q[1] = ap[2];
    // B (32x16 bf16): lanes 0-15 hold K=[kb..kb+7]+[kb+16..kb+23] of column n; lanes 16-31 K offset +8
    const u32x4* bp = (const u32x4*)(brow + kb);
    b.q[0] = bp[0];
    b.q[1] = bp[2];
    acc = __builtin_amdgcn_wmma_f32_16x16x32_bf16(false, a.v, false, b.v,
                                                  (short)0, acc, false, false);
  }
  return acc;
}

__global__ void coesn_setup(unsigned* w, int nwords) {
  int i = blockIdx.x * blockDim.x + threadIdx.x;
  if (i < nwords) w[i] = 0u;
}

__global__ __launch_bounds__(NTHR, 1) void coesn_persistent(
    const float* __restrict__ x, const float* __restrict__ x2h,
    const float* __restrict__ h2h, const float* __restrict__ bias,
    const float* __restrict__ lif2hrf, const float* __restrict__ gam,
    const float* __restrict__ eps, float* __restrict__ out,
    unsigned short* __restrict__ s_spk, unsigned short* __restrict__ lif_spk,
    unsigned* bar, float* cnts) {
  // exactly 64 KB LDS: both weight column-slices, transposed, bf16
  __shared__ __align__(16) unsigned short wt_h2h[COLS * Hs];
  __shared__ __align__(16) unsigned short wt_lif[COLS * Hs];

  const int tid = threadIdx.x;
  const int n0 = blockIdx.x * COLS;

  // one-time: transpose + convert this block's 16-column weight slices into LDS
  for (int i = tid; i < COLS * Hs; i += NTHR) {
    int n = i & (COLS - 1);
    int k = i >> 4;
    wt_h2h[n * Hs + k] = f2bf(h2h[(size_t)k * Hs + n0 + n]);
    wt_lif[n * Hs + k] = f2bf(lif2hrf[(size_t)k * Hs + n0 + n]);
  }
  __syncthreads();

  const int lane = tid & 31;
  const int wave = tid >> 5;
  const int half = lane >> 4;
  const int nl   = lane & 15;
  const int m0   = wave * 16;     // this wave's 16-row tile of B
  const int mA   = m0 + nl;       // A-fragment row loaded by this lane
  const int nG   = n0 + nl;       // global column owned by this lane's C/D fragment

  // per-column params live in registers (one column per lane)
  const float bias_r = bias[nG];
  const float x2h_r  = x2h[nG];
  const float gam_r  = gam[nG];
  const float eps_r  = eps[nG];
  const float ref_decay = expf(-DT_C / TAU_REF_C);

  // per-(m,n) state in registers: lane owns rows m0+8*half+r, r=0..7, column nG
  float hy[8] = {}, hz[8] = {}, lifv[8] = {}, refp[8] = {}, hysum[8] = {}, hysq[8] = {};
  float hrf_cnt = 0.f, lif_cnt = 0.f;

  for (int t = 0; t < Ls; ++t) {
    // ---- Phase A: cur = s @ h2h + x_t * x2h + bias ; LIF update -> lif spikes ----
    v8f acc = {0.f, 0.f, 0.f, 0.f, 0.f, 0.f, 0.f, 0.f};
    acc = spike_matmul(s_spk, wt_h2h, mA, half, nl, acc);
#pragma unroll
    for (int r = 0; r < 8; ++r) {
      const int m = m0 + 8 * half + r;
      const float xv = x[(size_t)m * Ls + t];
      const float cur = acc[r] + xv * x2h_r + bias_r;
      lifv[r] += DT_C * (-lifv[r] * (1.0f / LIF_TAU_M_C) + cur);
      const float sp = lifv[r] > THETA_LIF_C ? 1.0f : 0.0f;
      lifv[r] -= sp * THETA_LIF_C;
      lif_cnt += sp;
      lif_spk[(size_t)m * Hs + nG] =
          sp != 0.0f ? (unsigned short)0x3F80 : (unsigned short)0;  // bf16 1.0 / 0.0
    }
    grid_sync(bar, bar + 1);

    // ---- Phase B: drive = lif_s @ lif2hrf ; HRF oscillator update -> s spikes ----
    v8f dacc = {0.f, 0.f, 0.f, 0.f, 0.f, 0.f, 0.f, 0.f};
    dacc = spike_matmul(lif_spk, wt_lif, mA, half, nl, dacc);
#pragma unroll
    for (int r = 0; r < 8; ++r) {
      const int m = m0 + 8 * half + r;
      hz[r] += DT_C * (dacc[r] - gam_r * hy[r] - eps_r * hz[r]);
      hy[r] += DT_C * hz[r];
      const float sn = (hy[r] - THETA_RF_C - refp[r]) > 0.0f ? 1.0f : 0.0f;
      refp[r] = refp[r] * ref_decay + sn;
      hysum[r] += hy[r];
      hysq[r] += hy[r] * hy[r];
      hrf_cnt += sn;
      s_spk[(size_t)m * Hs + nG] =
          sn != 0.0f ? (unsigned short)0x3F80 : (unsigned short)0;
    }
    grid_sync(bar, bar + 1);
  }

  // ---- features: [hy_rms | hy_std | hy] per row, (B, 3H) row-major ----
  const float invL = 1.0f / (float)Ls;
#pragma unroll
  for (int r = 0; r < 8; ++r) {
    const int m = m0 + 8 * half + r;
    const float ms = hysq[r] * invL;
    const float mean = hysum[r] * invL;
    float var = ms - mean * mean;
    var = var < 1e-8f ? 1e-8f : var;
    float* row = out + (size_t)m * (3 * Hs);
    row[nG]          = sqrtf(ms + 1e-8f);
    row[Hs + nG]     = sqrtf(var);
    row[2 * Hs + nG] = hy[r];
  }
  atomicAdd(&cnts[0], hrf_cnt);
  atomicAdd(&cnts[1], lif_cnt);
  grid_sync(bar, bar + 1);
  if (blockIdx.x == 0 && tid == 0) {
    const float denom = (float)Bs * (float)Ls * (float)Hs;
    const float hn = __hip_atomic_load(&cnts[0], __ATOMIC_RELAXED, __HIP_MEMORY_SCOPE_AGENT);
    const float ln = __hip_atomic_load(&cnts[1], __ATOMIC_RELAXED, __HIP_MEMORY_SCOPE_AGENT);
    const size_t base = (size_t)Bs * 3 * Hs;
    out[base + 0] = hn / denom;   // r_total (== r_hrf, count_lif_spikes=False)
    out[base + 1] = hn / denom;   // r_hrf
    out[base + 2] = ln / denom;   // r_lif
  }
}

extern "C" void kernel_launch(void* const* d_in, const int* in_sizes, int n_in,
                              void* d_out, int out_size, void* d_ws, size_t ws_size,
                              hipStream_t stream) {
  (void)in_sizes; (void)n_in; (void)out_size; (void)ws_size;
  const float* x       = (const float*)d_in[0];
  const float* x2h     = (const float*)d_in[1];
  const float* h2h     = (const float*)d_in[2];
  const float* bias    = (const float*)d_in[3];
  const float* lif2hrf = (const float*)d_in[4];
  const float* gam     = (const float*)d_in[5];
  const float* eps     = (const float*)d_in[6];

  // workspace layout:
  //   [0      , 262144) : s_spk   (128x1024 bf16)
  //   [262144 , 524288) : lif_spk (128x1024 bf16)
  //   [524288 , 524296) : barrier {count, generation}
  //   [524296 , 524304) : float counters {hrf_n, lif_n}
  char* ws = (char*)d_ws;
  unsigned short* s_spk   = (unsigned short*)ws;
  unsigned short* lif_spk = (unsigned short*)(ws + 262144);
  unsigned*       bar     = (unsigned*)(ws + 524288);
  float*          cnts    = (float*)(ws + 524296);

  const int nwords = 524304 / 4;   // zero spikes + barrier + counters each launch
  coesn_setup<<<(nwords + 255) / 256, 256, 0, stream>>>((unsigned*)d_ws, nwords);
  coesn_persistent<<<NBLK, NTHR, 0, stream>>>(x, x2h, h2h, bias, lif2hrf, gam, eps,
                                              (float*)d_out, s_spk, lif_spk, bar, cnts);
}
